// AttentionHead_44117904064899
// MI455X (gfx1250) — compile-verified, matrix operand order checked by reading, not run on hardware
//
#include <hip/hip_runtime.h>

typedef __attribute__((ext_vector_type(16))) __bf16 v16bf;
typedef __attribute__((ext_vector_type(8)))  float  v8f;
typedef __attribute__((ext_vector_type(4)))  int    v4i;

#define D_MODEL 2048
#define N_HEADS 16
#define SEQ     64
#define BATCH   256
#define HD      128          // head dim
#define E3      6144         // 3*d_model
#define NTOK    (BATCH*SEQ)  // 16384

union FragBF { v16bf v; uint4 q[2]; };

// Load a 16x32 bf16 WMMA fragment (A layout; B uses the mirrored identical
// pattern) from a row-major [rows x stride] source (LDS or global).
// lane L reads row (L&15), two contiguous 16B chunks at k = 8*(L>>4) and 16+8*(L>>4).
__device__ __forceinline__ v16bf ldfrag(const __bf16* base, int stride, int lane) {
    const __bf16* p = base + (size_t)(lane & 15) * stride + ((lane >> 4) << 3);
    FragBF f;
    f.q[0] = *(const uint4*)(p);
    f.q[1] = *(const uint4*)(p + 16);
    return f.v;
}

__device__ __forceinline__ v8f wmma_bf16(v16bf a, v16bf b, v8f c) {
    // 8 args: (neg_a, A, neg_b, B, c_mod, C, reuse_a, reuse_b)
    return __builtin_amdgcn_wmma_f32_16x16x32_bf16(false, a, false, b, (short)0, c,
                                                   false, false);
}

// 16-byte global -> LDS copy. Prefer the CDNA5 async-to-LDS path (ASYNCcnt,
// no VGPR round-trip); fall back to a sync register copy if the builtin is
// absent on this toolchain.
#if defined(__has_builtin)
#  if __has_builtin(__builtin_amdgcn_global_load_async_to_lds_b128)
#    define HAVE_ASYNC_LDS 1
#  endif
#endif
#ifndef HAVE_ASYNC_LDS
#  define HAVE_ASYNC_LDS 0
#endif

typedef v4i __attribute__((address_space(1)))* gas_b128_t;  // global int4*
typedef v4i __attribute__((address_space(3)))* las_b128_t;  // LDS int4*

__device__ __forceinline__ void cp16_g2l(const __bf16* gsrc, __bf16* ldst) {
#if HAVE_ASYNC_LDS
    __builtin_amdgcn_global_load_async_to_lds_b128(
        (gas_b128_t)gsrc, (las_b128_t)ldst, 0, 0);
#else
    *(uint4*)ldst = *(const uint4*)gsrc;
#endif
}

__device__ __forceinline__ void wait_async0() {
#if defined(__has_builtin) && __has_builtin(__builtin_amdgcn_s_wait_asynccnt)
    __builtin_amdgcn_s_wait_asynccnt(0);
#else
    asm volatile("s_wait_asynccnt 0" ::: "memory");
#endif
}

// ---------------------------------------------------------------- converts
__global__ void cvt_f32_bf16(const float* __restrict__ in, __bf16* __restrict__ out, int n) {
    int i = blockIdx.x * blockDim.x + threadIdx.x;
    int stride = gridDim.x * blockDim.x;
    for (; i < n; i += stride) out[i] = (__bf16)in[i];
}

// ---------------------------------------------------------------- GEMM
// C[M,N] = A[M,K] * W[N,K]^T + bias[N];  A,W bf16 row-major; C bf16 or f32.
// Block tile 64x128, 8 waves -> each wave a 32x32 tile (2x2 WMMA accs).
// Double-buffered LDS fed by async global->LDS copies: iteration s issues the
// copies for tile s+1 into buf^1, runs WMMAs on buf, then s_wait_asynccnt 0 +
// barrier before swapping.
template <bool OUT_BF16>
__global__ __launch_bounds__(256) void gemm_wmma(
    const __bf16* __restrict__ A, const __bf16* __restrict__ W,
    const float* __restrict__ bias, void* __restrict__ Cout,
    int M, int N, int K)
{
    constexpr int BM = 64, BN = 128, BK = 64, PAD = 8;
    __shared__ __bf16 As[2][BM][BK + PAD];
    __shared__ __bf16 Bs[2][BN][BK + PAD];

    const int tid  = threadIdx.x;
    const int lane = tid & 31;
    const int wave = tid >> 5;
    const int wm   = wave >> 2;          // 0..1 : 32-row slice
    const int wn   = wave & 3;           // 0..3 : 32-col slice
    const int bm   = blockIdx.x * BM;
    const int bn   = blockIdx.y * BN;

    v8f acc00 = {}, acc01 = {}, acc10 = {}, acc11 = {};

    auto copy_tiles = [&](int buf, int k0) {
        #pragma unroll
        for (int i = tid; i < BM * (BK / 8); i += 256) {   // 2 iters
            int r = i >> 3, c = i & 7;
            cp16_g2l(&A[(size_t)(bm + r) * K + k0 + c * 8], &As[buf][r][c * 8]);
        }
        #pragma unroll
        for (int i = tid; i < BN * (BK / 8); i += 256) {   // 4 iters
            int r = i >> 3, c = i & 7;
            cp16_g2l(&W[(size_t)(bn + r) * K + k0 + c * 8], &Bs[buf][r][c * 8]);
        }
    };

    const int nsteps = K / BK;
    copy_tiles(0, 0);
    wait_async0();
    __syncthreads();

    for (int s = 0; s < nsteps; ++s) {
        const int cur = s & 1;
        if (s + 1 < nsteps) copy_tiles(cur ^ 1, (s + 1) * BK);

        #pragma unroll
        for (int kk = 0; kk < BK; kk += 32) {
            v16bf a0 = ldfrag(&As[cur][wm * 32     ][kk], BK + PAD, lane);
            v16bf a1 = ldfrag(&As[cur][wm * 32 + 16][kk], BK + PAD, lane);
            v16bf b0 = ldfrag(&Bs[cur][wn * 32     ][kk], BK + PAD, lane);
            v16bf b1 = ldfrag(&Bs[cur][wn * 32 + 16][kk], BK + PAD, lane);
            acc00 = wmma_bf16(a0, b0, acc00);
            acc01 = wmma_bf16(a0, b1, acc01);
            acc10 = wmma_bf16(a1, b0, acc10);
            acc11 = wmma_bf16(a1, b1, acc11);
        }

        wait_async0();
        __syncthreads();
    }

    // Epilogue: D tile layout = lane (L&15) -> col, VGPR g + 8*(L>>4) -> row
    const int n0 = lane & 15;
    const int r8 = (lane >> 4) * 8;
    #pragma unroll
    for (int tm = 0; tm < 2; ++tm) {
        #pragma unroll
        for (int tn = 0; tn < 2; ++tn) {
            v8f acc = (tm == 0) ? (tn == 0 ? acc00 : acc01)
                                : (tn == 0 ? acc10 : acc11);
            int col  = bn + wn * 32 + tn * 16 + n0;
            float bv = bias[col];
            #pragma unroll
            for (int g = 0; g < 8; ++g) {
                int row = bm + wm * 32 + tm * 16 + r8 + g;
                float v = acc[g] + bv;
                if (OUT_BF16)
                    ((__bf16*)Cout)[(size_t)row * N + col] = (__bf16)v;
                else
                    ((float*)Cout)[(size_t)row * N + col] = v;
            }
        }
    }
}

// ---------------------------------------------------------------- attention
// One block per (b,h). qkv layout: [token, 6144] with q|k|v at +0/+2048/+4096,
// head h occupying cols h*128..h*128+127. Writes bf16 out[token, 2048]
// already in [B,S,H*hd] order for the output projection.
__global__ __launch_bounds__(256) void attn_wmma(
    const __bf16* __restrict__ qkv, __bf16* __restrict__ outp)
{
    const int b = blockIdx.x;
    const int h = blockIdx.y;
    constexpr int S = SEQ;
    constexpr float SCALE = 0.02209708691207961f;  // 1/sqrt(d_model=2048)

    __shared__ __bf16 Vt[HD][S + 8];   // V transposed: Vt[d][j]
    __shared__ float  Sf[S][S + 1];    // raw scores
    __shared__ __bf16 Pb[S][S + 8];    // softmaxed probs (row-major [q][k])

    const int tid = threadIdx.x, lane = tid & 31, wave = tid >> 5;
    const size_t tok0 = (size_t)b * S;
    const __bf16* Qg = qkv + tok0 * E3 + (size_t)h * HD;
    const __bf16* Kg = Qg + D_MODEL;
    const __bf16* Vg = Qg + 2 * D_MODEL;

    // Stage V transposed into LDS (rows contiguous in global, scatter to LDS).
    for (int i = tid; i < S * (HD / 8); i += 256) {
        int j = i >> 4;    // seq row 0..63
        int c = i & 15;    // 8-elem chunk 0..15
        uint4 raw = *(const uint4*)(Vg + (size_t)j * E3 + c * 8);
        const __bf16* pv = (const __bf16*)&raw;
        #pragma unroll
        for (int e = 0; e < 8; ++e) Vt[c * 8 + e][j] = pv[e];
    }

    // Phase 1: S = Q K^T. 16 (4x4) 16x16 tiles over 8 waves: one tm, two tn each.
    // Q/K fragments are loaded straight from global (rows are 256B contiguous).
    {
        const int tm  = wave >> 1;
        const int tn0 = (wave & 1) * 2;
        v8f s0 = {}, s1 = {};
        #pragma unroll
        for (int kc = 0; kc < HD; kc += 32) {
            v16bf a  = ldfrag(Qg + (size_t)(tm * 16) * E3 + kc, E3, lane);
            v16bf b0 = ldfrag(Kg + (size_t)(tn0 * 16) * E3 + kc, E3, lane);
            v16bf b1 = ldfrag(Kg + (size_t)((tn0 + 1) * 16) * E3 + kc, E3, lane);
            s0 = wmma_bf16(a, b0, s0);
            s1 = wmma_bf16(a, b1, s1);
        }
        const int n0 = lane & 15, r8 = (lane >> 4) * 8;
        #pragma unroll
        for (int g = 0; g < 8; ++g) {
            Sf[tm * 16 + r8 + g][tn0 * 16 + n0]       = s0[g];
            Sf[tm * 16 + r8 + g][(tn0 + 1) * 16 + n0] = s1[g];
        }
    }
    __syncthreads();

    // Phase 2: causal mask + softmax over the QUERY axis (per key column k,
    // valid rows are q >= k, faithful to the reference's axis=-2 softmax).
    if (tid < S) {
        const int k = tid;
        float m = -3.0e38f;
        for (int q = k; q < S; ++q) m = fmaxf(m, Sf[q][k] * SCALE);
        float sum = 0.f;
        for (int q = k; q < S; ++q) sum += __expf(Sf[q][k] * SCALE - m);
        float inv = 1.f / sum;
        for (int q = 0; q < S; ++q) {
            float p = (q >= k) ? __expf(Sf[q][k] * SCALE - m) * inv : 0.f;
            Pb[q][k] = (__bf16)p;
        }
    }
    __syncthreads();

    // Phase 3: O = P @ V. O is 64x128 = 4(tm) x 8(tn) tiles; wave = tn.
    {
        const int n0 = lane & 15, r8 = (lane >> 4) * 8;
        #pragma unroll
        for (int tm = 0; tm < 4; ++tm) {
            v8f o = {};
            #pragma unroll
            for (int kc = 0; kc < S; kc += 32) {
                v16bf a  = ldfrag(&Pb[tm * 16][kc], S + 8, lane);
                v16bf bb = ldfrag(&Vt[wave * 16][kc], S + 8, lane);
                o = wmma_bf16(a, bb, o);
            }
            #pragma unroll
            for (int g = 0; g < 8; ++g) {
                int srow = tm * 16 + r8 + g;
                outp[(tok0 + srow) * (size_t)D_MODEL + (size_t)h * HD +
                     wave * 16 + n0] = (__bf16)o[g];
            }
        }
    }
}

// ---------------------------------------------------------------- launcher
extern "C" void kernel_launch(void* const* d_in, const int* in_sizes, int n_in,
                              void* d_out, int out_size, void* d_ws, size_t ws_size,
                              hipStream_t stream) {
    (void)in_sizes; (void)n_in; (void)out_size; (void)ws_size;
    const float* x     = (const float*)d_in[0];
    const float* W_qkv = (const float*)d_in[1];
    const float* b_qkv = (const float*)d_in[2];
    const float* W_out = (const float*)d_in[3];
    const float* b_out = (const float*)d_in[4];
    float* out = (float*)d_out;

    // Workspace carve (bf16): x | W_qkv | W_out | qkv | attn_out  (~369 MB)
    __bf16* xb    = (__bf16*)d_ws;
    __bf16* wqkvb = xb    + (size_t)NTOK * D_MODEL;
    __bf16* woutb = wqkvb + (size_t)E3 * D_MODEL;
    __bf16* qkvb  = woutb + (size_t)D_MODEL * D_MODEL;
    __bf16* attnb = qkvb  + (size_t)NTOK * E3;

    cvt_f32_bf16<<<2048, 256, 0, stream>>>(x,     xb,    NTOK * D_MODEL);
    cvt_f32_bf16<<<2048, 256, 0, stream>>>(W_qkv, wqkvb, E3 * D_MODEL);
    cvt_f32_bf16<<<1024, 256, 0, stream>>>(W_out, woutb, D_MODEL * D_MODEL);

    // QKV projection: [16384,6144] = x @ W_qkv^T + b_qkv   (bf16 out)
    gemm_wmma<true><<<dim3(NTOK / 64, E3 / 128), 256, 0, stream>>>(
        xb, wqkvb, b_qkv, qkvb, NTOK, E3, D_MODEL);

    // Attention per (b,h)
    attn_wmma<<<dim3(BATCH, N_HEADS), 256, 0, stream>>>(qkvb, attnb);

    // Output projection: [16384,2048] = attn @ W_out^T + b_out  (f32 out)
    gemm_wmma<false><<<dim3(NTOK / 64, D_MODEL / 128), 256, 0, stream>>>(
        attnb, woutb, b_out, out, NTOK, D_MODEL, D_MODEL);
}
